// TimeAwareRNN_44616120271295
// MI455X (gfx1250) — matrix operand order repaired and checked by reading, not compiled, not run on hardware
//
#include <hip/hip_runtime.h>
#include <hip/hip_bf16.h>

// ---------------- types ----------------
typedef __bf16 v16bf __attribute__((ext_vector_type(16)));
typedef float  v8f   __attribute__((ext_vector_type(8)));

#define LVAL 256
#define BVAL 512
#define KS   512

// fp32 -> bf16 (round to nearest even)
__device__ inline unsigned short f2bf(float f) {
  unsigned u = __float_as_uint(f);
  u += 0x7FFFu + ((u >> 16) & 1u);
  return (unsigned short)(u >> 16);
}

__device__ inline float sigmoidf_(float x) { return 1.f / (1.f + __expf(-x)); }

// fast tanh via v_exp_f32
__device__ inline float tanhf_(float x) {
  float e = __expf(-2.f * fabsf(x));
  float t = (1.f - e) / (1.f + e);
  return copysignf(t, x);
}

// ---------------- weight packing ----------------
// Pack W (N x 512, row-major, fp32) into bf16 WMMA B-fragments for
// D = A(16x32) x B(32x16): frag index = ntile*16 + ktile.
// B layout (16-bit, K=32): lanes 0-15 hold K=0..7,16..23, lanes 16-31
// hold K=8..15,24..31; N = n0 + lane%16. One lane = 16 bf16 = 32 contiguous bytes.
// Fragment ranges: Wx [0,1536) Wg [1536,2560) Wlin [2560,3072) Ly1 [3072,3584)
#define WX_OFF   0
#define WG_OFF   786432
#define WLIN_OFF 1310720
#define LY1_OFF  1572864

__global__ __launch_bounds__(256) void pack_weights_kernel(
    const float* __restrict__ Wx, const float* __restrict__ Wg,
    const float* __restrict__ Wlin, const float* __restrict__ Ly1,
    unsigned short* __restrict__ packed)
{
  int lane = threadIdx.x & 31;
  int f = blockIdx.x * 8 + (threadIdx.x >> 5);
  if (f >= 3584) return;
  const float* src; int lf; unsigned short* dst;
  if (f < 1536)      { src = Wx;   lf = f;        dst = packed + WX_OFF; }
  else if (f < 2560) { src = Wg;   lf = f - 1536; dst = packed + WG_OFF; }
  else if (f < 3072) { src = Wlin; lf = f - 2560; dst = packed + WLIN_OFF; }
  else               { src = Ly1;  lf = f - 3072; dst = packed + LY1_OFF; }
  int ntile = lf >> 4, ktile = lf & 15;
  int n = ntile * 16 + (lane & 15);
  int kbase = ktile * 32 + ((lane >= 16) ? 8 : 0);
  unsigned short* out = dst + ((size_t)lf * 32 + lane) * 16;
  const float* row = src + (size_t)n * KS + kbase;
#pragma unroll
  for (int e = 0; e < 16; ++e) out[e] = f2bf(row[(e < 8) ? e : e + 8]);
}

// ---------------- A-fragment helpers ----------------
// A layout (16-bit, 16x32): lanes 0-15: M=lane, e0..7 <-> K%32=0..7,
// e8..15 <-> 16..23; lanes 16-31: M=lane-16, e0..7 <-> 8..15, e8..15 <-> 24..31.
// Thread owns row m, ktile kt, 32 contiguous K values v[0..31].
__device__ inline void writeFragRow(unsigned short* frag, int kt, int m,
                                    const float* v) {
  unsigned short* p0 = frag + (kt * 32 + m) * 16;
  unsigned short* p1 = frag + (kt * 32 + 16 + m) * 16;
#pragma unroll
  for (int e = 0; e < 8; ++e) {
    p0[e]     = f2bf(v[e]);
    p0[8 + e] = f2bf(v[16 + e]);
    p1[e]     = f2bf(v[8 + e]);
    p1[8 + e] = f2bf(v[24 + e]);
  }
}

// NT adjacent 16x16 output tiles, K=512 fully accumulated.
// k-outer / tile-inner: one A-frag load feeds NT independent WMMA chains.
// boff is the wave-uniform element offset of the first tile inside `packedW`.
// readfirstlane makes it provably scalar (SGPR), then the empty volatile asm
// makes it opaque per call so the B-fragment global loads cannot be hoisted
// out of the 256-step time loop (and spilled to scratch): they stay as
// per-iteration saddr-form global_load_b128 hitting L2.
template <int NT>
__device__ inline void wmma_multi(const unsigned short* __restrict__ afrag,
                                  const unsigned short* __restrict__ packedW,
                                  unsigned boff, int lane, v8f* acc) {
  int so = __builtin_amdgcn_readfirstlane((int)boff);
  asm volatile("" : "+s"(so));
  const unsigned short* bb = packedW + (unsigned)so;
#pragma unroll
  for (int kt = 0; kt < 16; ++kt) {
    v16bf a = *(const v16bf*)(const void*)(afrag + (kt * 32 + lane) * 16);
#pragma unroll
    for (int i = 0; i < NT; ++i) {
      v16bf b = *(const v16bf*)(const void*)(bb + (size_t)i * 8192 +
                                             (kt * 32 + lane) * 16);
      acc[i] = __builtin_amdgcn_wmma_f32_16x16x32_bf16(false, a, false, b,
                                                       (short)0, acc[i],
                                                       false, false);
    }
  }
}

// ---------------- fused RK4 scan ----------------
// grid = 32 blocks (16 batch rows each), 256 threads = 8 waves.
#define STR 520                 // padded fp32 row stride (bank decorrelation)
#define OFF_HSF  0              // stage-h fp32 [16][STR]
#define OFF_ZBUF 33280          // z gate fp32 [16][STR]
#define OFF_FBUF 66560          // f / tanh buffer fp32 [16][STR]
#define OFF_HSA  99840          // stage-h A-frags bf16 [16][32][16]
#define OFF_RHA  116224         // (r*h) A-frags
#define OFF_XEA  132608         // xe A-frags
#define OFF_WIN  148992         // Win staged [512][16] fp32
#define OFF_BIN  181760         // bin staged [512] fp32
#define SMEM_BYTES 183808

__global__ __launch_bounds__(256, 1) void taw_rnn_scan_kernel(
    const float* __restrict__ seq, const float* __restrict__ state0,
    const float* __restrict__ Win, const float* __restrict__ bin,
    const float* __restrict__ bg, const float* __restrict__ blin,
    const float* __restrict__ Ly2,
    const unsigned short* __restrict__ packed,
    float* __restrict__ out)
{
  extern __shared__ char smem[];
  float* hsF  = (float*)(smem + OFF_HSF);
  float* zbuf = (float*)(smem + OFF_ZBUF);
  float* fbuf = (float*)(smem + OFF_FBUF);
  unsigned short* hsA = (unsigned short*)(smem + OFF_HSA);
  unsigned short* rhA = (unsigned short*)(smem + OFF_RHA);
  unsigned short* xeA = (unsigned short*)(smem + OFF_XEA);
  float* winS = (float*)(smem + OFF_WIN);
  float* binS = (float*)(smem + OFF_BIN);

  const unsigned short* pWx   = packed + WX_OFF;
  const unsigned short* pWg   = packed + WG_OFF;
  const unsigned short* pWlin = packed + WLIN_OFF;
  const unsigned short* pLy1  = packed + LY1_OFF;

  const int tid  = threadIdx.x;
  const int lane = tid & 31;
  const int wave = tid >> 5;
  const int b0   = blockIdx.x * 16;

  // elementwise role: row em, ktile ekt -> K range [ej0, ej0+32)
  const int em  = tid & 15;
  const int ekt = tid >> 4;
  const int ej0 = ekt * 32;

  // stage Win/bin once (re-read by every thread every step)
  for (int idx = tid; idx < 512 * 15; idx += 256) {
    int r = idx / 15, ci = idx - r * 15;
    winS[r * 16 + ci] = Win[idx];
  }
  for (int idx = tid; idx < 512; idx += 256) binS[idx] = bin[idx];

  // wave-role per-lane bias constants (N = tile_n0 + lane%16)
  float bgv[8], blinv[4];
#pragma unroll
  for (int i = 0; i < 8; ++i) bgv[i] = bg[(wave * 8 + i) * 16 + (lane & 15)];
#pragma unroll
  for (int j = 0; j < 4; ++j) blinv[j] = blin[(wave * 4 + j) * 16 + (lane & 15)];

  // persistent state in registers (this thread's 32 elements of h)
  float h[32], acc[32];
#pragma unroll
  for (int j = 0; j < 32; ++j) h[j] = state0[ej0 + j];
#pragma unroll
  for (int j = 0; j < 32; ++j) acc[j] = 0.f;

  __syncthreads();

  const v8f vzero = {};

  for (int t = 0; t < LVAL; ++t) {
    const float dtm =
        (t == 0) ? 0.f : seq[((size_t)(t - 1) * BVAL + b0 + em) * 16 + 15];

    // ---- A0: stage-h (= step-start h) fragments + hsF from registers ----
    writeFragRow(hsA, ekt, em, h);
#pragma unroll
    for (int j = 0; j < 32; ++j) hsF[em * STR + ej0 + j] = h[j];

    // ---- B: xe = tanh(x @ Win^T + bin) -> A-frags (VALU, K_IN=15) ----
    {
      float xr[15];
      const float* xp = seq + ((size_t)t * BVAL + b0 + em) * 16;
#pragma unroll
      for (int i = 0; i < 15; ++i) xr[i] = xp[i];
      float xe[32];
      for (int j = 0; j < 32; ++j) {
        int jj = ej0 + j;
        float a = binS[jj];
        const float* wr = winS + jj * 16;
#pragma unroll
        for (int i = 0; i < 15; ++i) a = fmaf(xr[i], wr[i], a);
        xe[j] = tanhf_(a);
      }
      writeFragRow(xeA, ekt, em, xe);
    }
    __syncthreads();

    // ---- C: readout GEMM tanh(h @ Ly1^T) and lx GEMMs (independent) ----
    {
      v8f cy[4];
#pragma unroll
      for (int j = 0; j < 4; ++j) cy[j] = vzero;
      wmma_multi<4>(hsA, pLy1, (unsigned)(wave * 4) * 8192u, lane, cy);
#pragma unroll
      for (int j = 0; j < 4; ++j) {
        int n = (wave * 4 + j) * 16 + (lane & 15);
        int mb = (lane >= 16) ? 8 : 0;
#pragma unroll
        for (int v = 0; v < 8; ++v) fbuf[(v + mb) * STR + n] = tanhf_(cy[j][v]);
      }
    }
    // lxg (Wx rows 0..1023) and lxl (rows 1024..1535), kept in registers
    v8f lxgF[8], lxlF[4];
#pragma unroll
    for (int i = 0; i < 8; ++i) lxgF[i] = vzero;
#pragma unroll
    for (int j = 0; j < 4; ++j) lxlF[j] = vzero;
    wmma_multi<8>(xeA, pWx, (unsigned)(wave * 8) * 8192u, lane, lxgF);
    wmma_multi<4>(xeA, pWx, (unsigned)(64 + wave * 4) * 8192u, lane, lxlF);
    __syncthreads();

    // ---- D: y_t = tanh(h@Ly1^T) @ Ly2^T  (N=8, 2 threads per output) ----
    {
      int m = tid >> 4;
      int n = (tid >> 1) & 7;
      int half = tid & 1;
      const float4* l2 = (const float4*)(Ly2 + (size_t)n * KS + half * 256);
      const float* fb = fbuf + m * STR + half * 256;
      float y = 0.f;
#pragma unroll 4
      for (int q = 0; q < 64; ++q) {
        float4 w = l2[q];
        float4 vv = *(const float4*)(fb + q * 4);
        y = fmaf(vv.x, w.x, fmaf(vv.y, w.y, fmaf(vv.z, w.z, fmaf(vv.w, w.w, y))));
      }
      y += __shfl_xor(y, 1);
      if (half == 0) out[((size_t)t * BVAL + b0 + m) * 8 + n] = y;
    }
    __syncthreads();

    // ---- E: four RK4 stages ----
#pragma unroll 1
    for (int s = 0; s < 4; ++s) {
      // E1: gates logit = lxg + hs@Wg^T + bg ; sigmoid.
      //     waves 0-3 own z (n<512) -> zbuf; waves 4-7 own r -> rhA=bf16(r*hs)
      {
        v8f c8[8];
#pragma unroll
        for (int i = 0; i < 8; ++i) c8[i] = lxgF[i];
        wmma_multi<8>(hsA, pWg, (unsigned)(wave * 8) * 8192u, lane, c8);
#pragma unroll
        for (int i = 0; i < 8; ++i) {
          int nt = wave * 8 + i;
          int n = nt * 16 + (lane & 15);
          int mb = (lane >= 16) ? 8 : 0;
          if (nt < 32) {  // z region (uniform per wave)
#pragma unroll
            for (int v = 0; v < 8; ++v)
              zbuf[(v + mb) * STR + n] = sigmoidf_(c8[i][v] + bgv[i]);
          } else {        // r region -> repack r*hs into A-fragment layout
            int kk = n - 512;
            int kt = kk >> 5, q = kk & 31;
            int lp, e;
            if (q < 8)       { lp = 0;  e = q;      }
            else if (q < 16) { lp = 16; e = q - 8;  }
            else if (q < 24) { lp = 0;  e = q - 8;  }
            else             { lp = 16; e = q - 16; }
#pragma unroll
            for (int v = 0; v < 8; ++v) {
              int mm = v + mb;
              float g  = sigmoidf_(c8[i][v] + bgv[i]);
              float rh = g * hsF[mm * STR + kk];
              rhA[(kt * 32 + lp + mm) * 16 + e] = f2bf(rh);
            }
          }
        }
      }
      __syncthreads();

      // E2: f = z * (tanh(lxl + (r*hs)@Wlin^T + blin) - hs)
      {
        v8f c4[4];
#pragma unroll
        for (int j = 0; j < 4; ++j) c4[j] = lxlF[j];
        wmma_multi<4>(rhA, pWlin, (unsigned)(wave * 4) * 8192u, lane, c4);
#pragma unroll
        for (int j = 0; j < 4; ++j) {
          int n = (wave * 4 + j) * 16 + (lane & 15);
          int mb = (lane >= 16) ? 8 : 0;
#pragma unroll
          for (int v = 0; v < 8; ++v) {
            int mm = v + mb;
            float tv = tanhf_(c4[j][v] + blinv[j]);
            fbuf[mm * STR + n] = zbuf[mm * STR + n] * (tv - hsF[mm * STR + n]);
          }
        }
      }
      __syncthreads();

      // E3: elementwise RK update (thread-private 32 elements)
      {
        float fv[32];
#pragma unroll
        for (int j = 0; j < 32; ++j) fv[j] = fbuf[em * STR + ej0 + j];
        float coef = (s == 1 || s == 2) ? 2.f : 1.f;
#pragma unroll
        for (int j = 0; j < 32; ++j)
          acc[j] = (s == 0) ? fv[j] : fmaf(coef, fv[j], acc[j]);
        if (s < 3) {
          float sc = (s == 2) ? dtm : 0.5f * dtm;
          float hn[32];
#pragma unroll
          for (int j = 0; j < 32; ++j) hn[j] = fmaf(sc, fv[j], h[j]);
#pragma unroll
          for (int j = 0; j < 32; ++j) hsF[em * STR + ej0 + j] = hn[j];
          writeFragRow(hsA, ekt, em, hn);
        } else {
#pragma unroll
          for (int j = 0; j < 32; ++j)
            h[j] = fmaf(dtm * (1.f / 6.f), acc[j], h[j]);
        }
      }
      __syncthreads();
    }
  }
}

// ---------------- launch ----------------
extern "C" void kernel_launch(void* const* d_in, const int* in_sizes, int n_in,
                              void* d_out, int out_size, void* d_ws,
                              size_t ws_size, hipStream_t stream) {
  const float* seq    = (const float*)d_in[0];
  const float* state0 = (const float*)d_in[1];
  const float* Win    = (const float*)d_in[2];
  const float* bin    = (const float*)d_in[3];
  const float* Wx     = (const float*)d_in[4];
  const float* Wg     = (const float*)d_in[5];
  const float* bg     = (const float*)d_in[6];
  const float* Wlin   = (const float*)d_in[7];
  const float* blin   = (const float*)d_in[8];
  const float* Ly1    = (const float*)d_in[9];
  const float* Ly2    = (const float*)d_in[10];

  unsigned short* packed = (unsigned short*)d_ws;  // 3.5 MB of bf16 fragments

  pack_weights_kernel<<<448, 256, 0, stream>>>(Wx, Wg, Wlin, Ly1, packed);
  taw_rnn_scan_kernel<<<32, 256, SMEM_BYTES, stream>>>(
      seq, state0, Win, bin, bg, blin, Ly2, packed, (float*)d_out);
}
